// SparseLinear_17927193493896
// MI455X (gfx1250) — matrix operand order repaired and checked by reading, not compiled, not run on hardware
//
#include <hip/hip_runtime.h>

typedef __attribute__((ext_vector_type(16))) __bf16 v16bf;
typedef __attribute__((ext_vector_type(2)))  __bf16 v2bf;
typedef __attribute__((ext_vector_type(8)))  float  v8f;

union Frag {
    uint4 q[2];
    v16bf v;
};

#define TILE_M 128
#define TILE_N 256
#define TILE_K 32
#define LDSTR  40            // padded LDS row stride in halves (32 + 8) -> conflict-free
#define K_DIM  4096
#define N_DIM  4096
#define M_DIM  8192

// two fp32 -> packed bf16 dword (lowers to v_cvt_pk_bf16_f32, RNE)
__device__ __forceinline__ unsigned pack_bf16(float a, float b) {
    union { v2bf h; unsigned u; } cvt;
    cvt.h.x = (__bf16)a;
    cvt.h.y = (__bf16)b;
    return cvt.u;
}

__global__ __launch_bounds__(256)
void sparse_linear_bf16_wmma(const float* __restrict__ x,
                             const float* __restrict__ w,
                             const float* __restrict__ bias,
                             float* __restrict__ out) {
    // double-buffered bf16 tiles: A[m][k], B[n][k] (transposed W tile)
    __shared__ __align__(16) unsigned short ldsA[2][TILE_M * LDSTR]; // 2 * 10 KB
    __shared__ __align__(16) unsigned short ldsB[2][TILE_N * LDSTR]; // 2 * 20 KB

    const int tid   = threadIdx.x;
    const int lane  = tid & 31;
    const int wid   = tid >> 5;        // 0..7
    const int waveM = wid >> 2;        // 0..1  -> 64-row band
    const int waveN = wid & 3;         // 0..3  -> 64-col band
    const int l15   = lane & 15;
    const int lhalf = lane >> 4;       // 0 or 1

    const int mBase = blockIdx.y * TILE_M;
    const int nBase = blockIdx.x * TILE_N;

    // staging coordinates (256 threads)
    const int ar = tid >> 3;           // 0..31 : x row within tile (step 32)
    const int ac = tid & 7;            // 0..7  : x float4-column in K
    const int br = tid >> 6;           // 0..3  : w even-k-row selector
    const int bc = tid & 63;           // 0..63 : w float4-column in N

    // uniform (SGPR) bases; per-thread offsets are 32-bit
    const float* xg = x + (size_t)mBase * K_DIM;
    const float* wg = w + nBase;

    const unsigned xoff = (unsigned)(ar * K_DIM + ac * 4);
    const unsigned woff = (unsigned)(bc * 4);

    v8f acc[4][4];
    #pragma unroll
    for (int mt = 0; mt < 4; ++mt)
        #pragma unroll
        for (int nt = 0; nt < 4; ++nt) {
            v8f z = {};
            acc[mt][nt] = z;
        }

    float4 xr[4], wr0[4], wr1[4];

#define LOAD_GLOBAL(kc)                                                               \
    {                                                                                 \
        const float* xgk = xg + (kc);                /* uniform s_add */              \
        const float* wgk = wg + (size_t)(kc) * N_DIM;                                 \
        _Pragma("unroll")                                                             \
        for (int r = 0; r < 4; ++r)                                                   \
            xr[r] = *(const float4*)(xgk + (xoff + (unsigned)(32 * r * K_DIM)));      \
        _Pragma("unroll")                                                             \
        for (int r = 0; r < 4; ++r) {                                                 \
            unsigned k0 = (unsigned)(8 * r + 2 * br);                                 \
            wr0[r] = *(const float4*)(wgk + (k0 * N_DIM + woff));                     \
            wr1[r] = *(const float4*)(wgk + ((k0 + 1) * N_DIM + woff));               \
        }                                                                             \
    }

#define STORE_LDS(p)                                                                  \
    {                                                                                 \
        _Pragma("unroll")                                                             \
        for (int r = 0; r < 4; ++r) {                                                 \
            uint2 q;                                                                  \
            q.x = pack_bf16(xr[r].x, xr[r].y);                                        \
            q.y = pack_bf16(xr[r].z, xr[r].w);                                        \
            *(uint2*)(&ldsA[p][(ar + 32 * r) * LDSTR + ac * 4]) = q;                  \
        }                                                                             \
        _Pragma("unroll")                                                             \
        for (int r = 0; r < 4; ++r) {                                                 \
            int k0 = 8 * r + 2 * br;  /* even */                                      \
            *(unsigned*)(&ldsB[p][(bc * 4 + 0) * LDSTR + k0]) =                       \
                pack_bf16(wr0[r].x, wr1[r].x);                                        \
            *(unsigned*)(&ldsB[p][(bc * 4 + 1) * LDSTR + k0]) =                       \
                pack_bf16(wr0[r].y, wr1[r].y);                                        \
            *(unsigned*)(&ldsB[p][(bc * 4 + 2) * LDSTR + k0]) =                       \
                pack_bf16(wr0[r].z, wr1[r].z);                                        \
            *(unsigned*)(&ldsB[p][(bc * 4 + 3) * LDSTR + k0]) =                       \
                pack_bf16(wr0[r].w, wr1[r].w);                                        \
        }                                                                             \
    }

#define COMPUTE(p)                                                                    \
    {                                                                                 \
        Frag bfrag[4];                                                                \
        _Pragma("unroll")                                                             \
        for (int nt = 0; nt < 4; ++nt) {                                              \
            int n = waveN * 64 + nt * 16 + l15;                                       \
            const uint4* q = (const uint4*)&ldsB[p][n * LDSTR + lhalf * 8];           \
            bfrag[nt].q[0] = q[0];                                                    \
            bfrag[nt].q[1] = q[2];                                                    \
        }                                                                             \
        Frag afrag[4];                                                                \
        _Pragma("unroll")                                                             \
        for (int mt = 0; mt < 4; ++mt) {                                              \
            int m = waveM * 64 + mt * 16 + l15;                                       \
            const unsigned short* base = &ldsA[p][m * LDSTR + lhalf * 8];             \
            afrag[mt].q[0] = *(const uint4*)(base);                                   \
            afrag[mt].q[1] = *(const uint4*)(base + 16);                              \
        }                                                                             \
        _Pragma("unroll")                                                             \
        for (int mt = 0; mt < 4; ++mt)                                                \
            _Pragma("unroll")                                                         \
            for (int nt = 0; nt < 4; ++nt)                                            \
                acc[mt][nt] = __builtin_amdgcn_wmma_f32_16x16x32_bf16(                \
                    false, afrag[mt].v, false, bfrag[nt].v, (short)0,                 \
                    acc[mt][nt], false, false);                                       \
    }

    // pipeline: stage chunk 0 -> buf0; each iteration computes from one buffer
    // while staging the next chunk into the other (single barrier per chunk).
    LOAD_GLOBAL(0);
    STORE_LDS(0);
    __syncthreads();

    for (int kc = TILE_K; kc < K_DIM - 2 * TILE_K; kc += 2 * TILE_K) {
        LOAD_GLOBAL(kc);
        COMPUTE(0);
        STORE_LDS(1);
        __syncthreads();

        LOAD_GLOBAL(kc + TILE_K);
        COMPUTE(1);
        STORE_LDS(0);
        __syncthreads();
    }
    // tail: chunk K-2 is in buf0; load+stage final chunk K-1, then drain
    LOAD_GLOBAL(K_DIM - TILE_K);
    COMPUTE(0);
    STORE_LDS(1);
    __syncthreads();
    COMPUTE(1);

    // epilogue: bias add + fp32 store (lanes 0..15 cols n..n+15, contiguous 64B)
    #pragma unroll
    for (int nt = 0; nt < 4; ++nt) {
        int col = nBase + waveN * 64 + nt * 16 + l15;
        float bv = bias[col];
        #pragma unroll
        for (int mt = 0; mt < 4; ++mt) {
            int row0 = mBase + waveM * 64 + mt * 16 + lhalf * 8;
            #pragma unroll
            for (int r = 0; r < 8; ++r)
                out[(size_t)(row0 + r) * N_DIM + col] = acc[mt][nt][r] + bv;
        }
    }
#undef LOAD_GLOBAL
#undef STORE_LDS
#undef COMPUTE
}

extern "C" void kernel_launch(void* const* d_in, const int* in_sizes, int n_in,
                              void* d_out, int out_size, void* d_ws, size_t ws_size,
                              hipStream_t stream) {
    (void)in_sizes; (void)n_in; (void)out_size; (void)d_ws; (void)ws_size;
    const float* x    = (const float*)d_in[0];
    const float* w    = (const float*)d_in[1];
    const float* bias = (const float*)d_in[2];
    float* out        = (float*)d_out;

    dim3 grid(N_DIM / TILE_N, M_DIM / TILE_M);  // (16, 64)
    dim3 block(256);                             // 8 wave32 waves
    sparse_linear_bf16_wmma<<<grid, block, 0, stream>>>(x, w, bias, out);
}
